// NNmodel_35227321762106
// MI455X (gfx1250) — matrix operand order repaired and checked by reading, not compiled, not run on hardware
//
#include <hip/hip_runtime.h>
#include <math.h>

// ---------------------------------------------------------------------------
// GNN (GCN + ECC branches, sum-pool, MLP head) for MI455X / gfx1250.
// Dense transforms: wave32 WMMA GEMM, fully padded operands (straight-line
// b128 loads), B tiles staged once per K-chunk through LDS and reused by
// 2 M-tile accumulators per wave. Sparse aggregation: global_atomic_add_f32.
// ---------------------------------------------------------------------------

typedef __attribute__((ext_vector_type(16))) _Float16 v16h;
typedef __attribute__((ext_vector_type(8)))  float    v8f;

union HV16 { v16h v; _Float16 h[16]; };
union FV8  { v8f  v; float    f[8];  };

#define N_NODES 60000
#define N_EDGES 240000
#define F_IN    10
#define C_W     16
#define G_B     512

static inline int ceil_div(int a, int b) { return (a + b - 1) / b; }

// ---------------------------------------------------------------------------
// WMMA GEMM on padded operands:
//   Out[M, *] = act(A[M, Kpad] @ B[Kpad, >=16*gridDim.y] + bias)
// Caller guarantees: M % (16*MTILES) == 0, Kpad % 32 == 0, lda == Kpad,
// A/B 16B aligned, B and bias padded to full 16-wide column tiles.
// One wave per (16*MTILES)x16 output strip: B tile staged in LDS once per
// K-chunk and reused by MTILES accumulators (back-to-back v_wmma).
// ACT: 0=none, 1=relu, 2=sigmoid. Stores predicated only on ncol < NcStore.
// ---------------------------------------------------------------------------
template <int ACT, int MTILES>
__global__ __launch_bounds__(32)
void gemm_wmma(const float* __restrict__ A, int lda,
               const float* __restrict__ B, int ldb,
               const float* __restrict__ bias,
               float* __restrict__ Out, int ldo,
               int Kpad, int NcStore) {
  __shared__ _Float16 Bs[32 * 16];

  const int l      = threadIdx.x;            // 0..31 (one full wave)
  const int mt0    = blockIdx.x * (16 * MTILES);
  const int nt     = blockIdx.y * 16;
  const int lane15 = l & 15;
  const int hiHalf = l >> 4;                 // 0: lanes 0-15, 1: lanes 16-31
  const int ncol   = nt + lane15;

  FV8 acc[MTILES] = {};

  for (int k0 = 0; k0 < Kpad; k0 += 32) {
    // ---- Stage B 32x16 tile into LDS: 4 float4 loads/lane, packed f16 stores
#pragma unroll
    for (int j = 0; j < 4; ++j) {
      const int idx = l + 32 * j;            // 0..127 float4-granules
      const int kr  = idx >> 2;              // B row 0..31
      const int c4  = (idx & 3) * 4;         // col group {0,4,8,12}
      const float4 bv = *(const float4*)(B + (size_t)(k0 + kr) * ldb + nt + c4);
      _Float16* d = &Bs[kr * 16 + c4];
      d[0] = (_Float16)bv.x; d[1] = (_Float16)bv.y;
      d[2] = (_Float16)bv.z; d[3] = (_Float16)bv.w;
    }
    __syncthreads();

    // ---- B fragment from LDS: lane owns column ncol; h -> k = hiHalf*16 + h
    HV16 b;
#pragma unroll
    for (int h = 0; h < 16; ++h)
      b.h[h] = Bs[(hiHalf * 16 + h) * 16 + lane15];

    // ---- MTILES A fragments + WMMAs against the shared B fragment
#pragma unroll
    for (int t = 0; t < MTILES; ++t) {
      const int mrow = mt0 + t * 16 + lane15;
      // h 0..15 -> k0+hiHalf*16+h
      const float4* Arow = (const float4*)(A + (size_t)mrow * lda + k0);
      const float4 a0 = Arow[hiHalf * 4 + 0];
      const float4 a1 = Arow[hiHalf * 4 + 1];
      const float4 a2 = Arow[hiHalf * 4 + 2];
      const float4 a3 = Arow[hiHalf * 4 + 3];
      HV16 a;
      a.h[ 0] = (_Float16)a0.x; a.h[ 1] = (_Float16)a0.y;
      a.h[ 2] = (_Float16)a0.z; a.h[ 3] = (_Float16)a0.w;
      a.h[ 4] = (_Float16)a1.x; a.h[ 5] = (_Float16)a1.y;
      a.h[ 6] = (_Float16)a1.z; a.h[ 7] = (_Float16)a1.w;
      a.h[ 8] = (_Float16)a2.x; a.h[ 9] = (_Float16)a2.y;
      a.h[10] = (_Float16)a2.z; a.h[11] = (_Float16)a2.w;
      a.h[12] = (_Float16)a3.x; a.h[13] = (_Float16)a3.y;
      a.h[14] = (_Float16)a3.z; a.h[15] = (_Float16)a3.w;

      acc[t].v = __builtin_amdgcn_wmma_f32_16x16x32_f16(
          /*neg_a=*/false, a.v, /*neg_b=*/false, b.v,
          /*c_mod=*/(short)0, acc[t].v, /*reuse_a=*/false, /*reuse_b=*/false);
    }
    __syncthreads();
  }

  const float bcol = (bias != nullptr) ? bias[ncol] : 0.f;
  if (ncol < NcStore) {
    // D layout: VGPR r, lanes 0-15 -> row mt+r; lanes 16-31 -> row mt+8+r
#pragma unroll
    for (int t = 0; t < MTILES; ++t) {
#pragma unroll
      for (int r = 0; r < 8; ++r) {
        const int row = mt0 + t * 16 + hiHalf * 8 + r;
        float v = acc[t].f[r] + bcol;
        if (ACT == 1) v = fmaxf(v, 0.f);
        if (ACT == 2) v = __builtin_amdgcn_rcpf(1.f + __expf(-v));
        Out[(size_t)row * ldo + ncol] = v;
      }
    }
  }
}

// ---------------------------------------------------------------------------
// Elementwise / repack / scatter kernels
// ---------------------------------------------------------------------------
__global__ void zero_f32(float* __restrict__ p, int n) {
  int i = blockIdx.x * blockDim.x + threadIdx.x;
  if (i < n) p[i] = 0.f;
}

// dst[drows, dcols] = src[srows, scols] zero-padded
__global__ void pad2d(const float* __restrict__ src, int srows, int scols,
                      float* __restrict__ dst, int drows, int dcols) {
  int i = blockIdx.x * blockDim.x + threadIdx.x;
  if (i >= drows * dcols) return;
  int r = i / dcols, c = i % dcols;
  dst[i] = (r < srows && c < scols) ? src[r * scols + c] : 0.f;
}

// WB1P [32 x 96] = rows 0..9: [W_gcn1 | M0 | M1 | M2 | Be1 | root1]; rows 10..31: 0
__global__ void repack_wb1(const float* __restrict__ Wg, const float* __restrict__ We,
                           const float* __restrict__ be, const float* __restrict__ root,
                           float* __restrict__ WB) {
  int i = blockIdx.x * blockDim.x + threadIdx.x;
  if (i >= 32 * 96) return;
  int f = i / 96, c = i % 96;
  float v = 0.f;
  if (f < F_IN) {
    if (c < 16)      v = Wg[f * 16 + c];
    else if (c < 64) { int s = (c - 16) / 16, o = (c - 16) % 16; v = We[s * (F_IN * 16) + f * 16 + o]; }
    else if (c < 80) { int o = c - 64; v = be[f * 16 + o]; }
    else             { int o = c - 80; v = root[f * 16 + o]; }
  }
  WB[i] = v;
}

// WB2P [32 x 160] = rows 0..15: [M0 | M1 | M2 | Be2 | root2]; rows 16..31: 0
__global__ void repack_wb2(const float* __restrict__ We, const float* __restrict__ be,
                           const float* __restrict__ root, float* __restrict__ WB) {
  int i = blockIdx.x * blockDim.x + threadIdx.x;
  if (i >= 32 * 160) return;
  int f = i / 160, c = i % 160;
  float v = 0.f;
  if (f < C_W) {
    if (c < 96)       { int s = c / 32, o = c % 32; v = We[s * (C_W * 32) + f * 32 + o]; }
    else if (c < 128) { int o = c - 96;  v = be[f * 32 + o]; }
    else              { int o = c - 128; v = root[f * 32 + o]; }
  }
  WB[i] = v;
}

// GCN aggregation: agg[rows[e]*ldagg + c] += a_vals[e] * Y[cols[e]*ldy + ycol0 + c]
__global__ void spmm_scatter(const int* __restrict__ ei, const float* __restrict__ vals,
                             const float* __restrict__ Y, int ldy, int ycol0, int Cc,
                             float* __restrict__ agg, int ldagg) {
  int i = blockIdx.x * blockDim.x + threadIdx.x;
  if (i >= N_EDGES * Cc) return;
  int e = i / Cc, c = i % Cc;
  int r   = ei[e];
  int col = ei[N_EDGES + e];
  atomicAdd(&agg[(size_t)r * ldagg + c], vals[e] * Y[(size_t)col * ldy + ycol0 + c]);
}

// ECC aggregation with U = X @ [M0|M1|M2|Be|...]:
//   m = e0*U[col,o] + e1*U[col,O+o] + e2*U[col,2O+o] + U[col,3O+o]
__global__ void ecc_scatter(const int* __restrict__ ei, const float* __restrict__ ef,
                            const float* __restrict__ U, int ldu, int ucol0, int O,
                            float* __restrict__ agg, int ldagg) {
  int i = blockIdx.x * blockDim.x + threadIdx.x;
  if (i >= N_EDGES * O) return;
  int e = i / O, o = i % O;
  int r   = ei[e];
  int col = ei[N_EDGES + e];
  const float* u = U + (size_t)col * ldu + ucol0;
  float e0 = ef[e * 3 + 0], e1 = ef[e * 3 + 1], e2 = ef[e * 3 + 2];
  float m = e0 * u[o] + e1 * u[O + o] + e2 * u[2 * O + o] + u[3 * O + o];
  atomicAdd(&agg[(size_t)r * ldagg + o], m);
}

// In-place: X[r*ldx + c] = relu(X + bias[c] (+ extra[r*ldext + extcol0 + c]))
__global__ void add_bias_relu(float* __restrict__ X, int ldx,
                              const float* __restrict__ extra, int ldext, int extcol0,
                              const float* __restrict__ bias, int n, int Cc) {
  int i = blockIdx.x * blockDim.x + threadIdx.x;
  if (i >= n * Cc) return;
  int r = i / Cc, c = i % Cc;
  float v = X[(size_t)r * ldx + c] + bias[c];
  if (extra != nullptr) v += extra[(size_t)r * ldext + extcol0 + c];
  X[(size_t)r * ldx + c] = v > 0.f ? v : 0.f;
}

// Per-graph sum pool into concatenated head input HC[G,64]
__global__ void pool_concat(const float* __restrict__ g2, const float* __restrict__ c2,
                            const int* __restrict__ seg, float* __restrict__ HC) {
  int i = blockIdx.x * blockDim.x + threadIdx.x;
  if (i >= N_NODES * 32) return;
  int node = i / 32, c = i % 32;
  int g = seg[node];
  atomicAdd(&HC[g * 64 + c],      g2[i]);
  atomicAdd(&HC[g * 64 + 32 + c], c2[i]);
}

// ---------------------------------------------------------------------------
extern "C" void kernel_launch(void* const* d_in, const int* in_sizes, int n_in,
                              void* d_out, int out_size, void* d_ws, size_t ws_size,
                              hipStream_t stream) {
  (void)in_sizes; (void)n_in; (void)out_size; (void)ws_size;

  const float* x       = (const float*)d_in[0];
  const float* a_vals  = (const float*)d_in[1];
  const float* e_feat  = (const float*)d_in[2];
  const int*   ei      = (const int*)  d_in[3];
  const int*   seg     = (const int*)  d_in[4];
  const float* W_gcn1  = (const float*)d_in[5];
  const float* b_gcn1  = (const float*)d_in[6];
  const float* W_gcn2  = (const float*)d_in[7];
  const float* b_gcn2  = (const float*)d_in[8];
  const float* We1     = (const float*)d_in[9];
  const float* be1     = (const float*)d_in[10];
  const float* root1   = (const float*)d_in[11];
  const float* bias1   = (const float*)d_in[12];
  const float* We2     = (const float*)d_in[13];
  const float* be2     = (const float*)d_in[14];
  const float* root2   = (const float*)d_in[15];
  const float* bias2   = (const float*)d_in[16];
  const float* Wd1     = (const float*)d_in[17];
  const float* bd1     = (const float*)d_in[18];
  const float* Wd2     = (const float*)d_in[19];
  const float* bd2     = (const float*)d_in[20];
  const float* Wo      = (const float*)d_in[21];
  const float* bo      = (const float*)d_in[22];
  float*       out     = (float*)d_out;
  float*       ws      = (float*)d_ws;

  // Workspace layout (floats); every block size is a multiple of 16 floats so
  // all GEMM operands stay 16-byte aligned for float4 loads. ~108 MB total.
  size_t off = 0;
  const size_t XP    = off; off += (size_t)N_NODES * 32;   // X padded K 10->32
  const size_t WB1P  = off; off += 32 * 96;                // packed L1 weights
  const size_t WGC2P = off; off += 32 * 32;                // W_gcn2 padded
  const size_t WB2P  = off; off += 32 * 160;               // packed L2 ECC weights
  const size_t WD1P  = off; off += 64 * 32;                // Wd1 padded cols
  const size_t BD1P  = off; off += 32;
  const size_t WD2P  = off; off += 32 * 32;                // Wd2 padded rows+cols
  const size_t BD2P  = off; off += 32;
  const size_t WOP   = off; off += 32 * 16;                // Wo padded
  const size_t BOP   = off; off += 16;
  const size_t XU1   = off; off += (size_t)N_NODES * 96;   // XP @ WB1P
  const size_t AG1   = off; off += (size_t)N_NODES * 32;   // GCN1 agg -> g1 (cols 0..15)
  const size_t GW2   = off; off += (size_t)N_NODES * 32;   // g1 @ W_gcn2
  const size_t AG2   = off; off += (size_t)N_NODES * 32;   // GCN2 agg -> g2
  const size_t AC1   = off; off += (size_t)N_NODES * 32;   // ECC1 agg -> c1 (cols 0..15)
  const size_t CU2   = off; off += (size_t)N_NODES * 160;  // c1 @ WB2P
  const size_t AC2   = off; off += (size_t)N_NODES * 32;   // ECC2 agg -> c2
  const size_t HC    = off; off += (size_t)G_B * 64;       // pooled concat [G,64]
  const size_t H1    = off; off += (size_t)G_B * 32;       // head hidden 1 (cols 16..31 = 0)
  const size_t H2    = off; off += (size_t)G_B * 32;       // head hidden 2 (cols 8..31 = 0)

  const int TB = 256;

  // ---- Pad / repack all GEMM operands (zero-fill pads)
  pad2d<<<ceil_div(N_NODES * 32, TB), TB, 0, stream>>>(x, N_NODES, F_IN, ws + XP, N_NODES, 32);
  repack_wb1<<<ceil_div(32 * 96, TB), TB, 0, stream>>>(W_gcn1, We1, be1, root1, ws + WB1P);
  repack_wb2<<<ceil_div(32 * 160, TB), TB, 0, stream>>>(We2, be2, root2, ws + WB2P);
  pad2d<<<ceil_div(32 * 32, TB), TB, 0, stream>>>(W_gcn2, 16, 32, ws + WGC2P, 32, 32);
  pad2d<<<ceil_div(64 * 32, TB), TB, 0, stream>>>(Wd1, 64, 16, ws + WD1P, 64, 32);
  pad2d<<<1, 32, 0, stream>>>(bd1, 1, 16, ws + BD1P, 1, 32);
  pad2d<<<ceil_div(32 * 32, TB), TB, 0, stream>>>(Wd2, 16, 8, ws + WD2P, 32, 32);
  pad2d<<<1, 32, 0, stream>>>(bd2, 1, 8, ws + BD2P, 1, 32);
  pad2d<<<ceil_div(32 * 16, TB), TB, 0, stream>>>(Wo, 8, 1, ws + WOP, 32, 16);
  pad2d<<<1, 16, 0, stream>>>(bo, 1, 1, ws + BOP, 1, 16);

  // ---- Zero all atomic-accumulation buffers (pad cols must stay zero)
  zero_f32<<<ceil_div(N_NODES * 32, TB), TB, 0, stream>>>(ws + AG1, N_NODES * 32);
  zero_f32<<<ceil_div(N_NODES * 32, TB), TB, 0, stream>>>(ws + AG2, N_NODES * 32);
  zero_f32<<<ceil_div(N_NODES * 32, TB), TB, 0, stream>>>(ws + AC1, N_NODES * 32);
  zero_f32<<<ceil_div(N_NODES * 32, TB), TB, 0, stream>>>(ws + AC2, N_NODES * 32);
  zero_f32<<<ceil_div(G_B * 64, TB),    TB, 0, stream>>>(ws + HC,  G_B * 64);

  // ---- Layer 1 fused GEMM: XU1 = XP[N,32] @ WB1P[32,96]
  gemm_wmma<0, 2><<<dim3(N_NODES / 32, 96 / 16), 32, 0, stream>>>(
      ws + XP, 32, ws + WB1P, 96, nullptr, ws + XU1, 96, 32, 96);

  // GCN1: agg = A @ (X W_gcn1); g1 = relu(agg + b)
  spmm_scatter<<<ceil_div(N_EDGES * 16, TB), TB, 0, stream>>>(
      ei, a_vals, ws + XU1, 96, 0, 16, ws + AG1, 32);
  add_bias_relu<<<ceil_div(N_NODES * 16, TB), TB, 0, stream>>>(
      ws + AG1, 32, nullptr, 0, 0, b_gcn1, N_NODES, 16);

  // ECC1: messages from U cols [16..80), root term cols [80..96)
  ecc_scatter<<<ceil_div(N_EDGES * 16, TB), TB, 0, stream>>>(
      ei, e_feat, ws + XU1, 96, 16, 16, ws + AC1, 32);
  add_bias_relu<<<ceil_div(N_NODES * 16, TB), TB, 0, stream>>>(
      ws + AC1, 32, ws + XU1, 96, 80, bias1, N_NODES, 16);

  // ---- Layer 2 GCN: GW2 = g1[N,32pad] @ WGC2P[32,32]
  gemm_wmma<0, 2><<<dim3(N_NODES / 32, 32 / 16), 32, 0, stream>>>(
      ws + AG1, 32, ws + WGC2P, 32, nullptr, ws + GW2, 32, 32, 32);
  spmm_scatter<<<ceil_div(N_EDGES * 32, TB), TB, 0, stream>>>(
      ei, a_vals, ws + GW2, 32, 0, 32, ws + AG2, 32);
  add_bias_relu<<<ceil_div(N_NODES * 32, TB), TB, 0, stream>>>(
      ws + AG2, 32, nullptr, 0, 0, b_gcn2, N_NODES, 32);

  // ---- Layer 2 ECC: CU2 = c1[N,32pad] @ WB2P[32,160]
  gemm_wmma<0, 2><<<dim3(N_NODES / 32, 160 / 16), 32, 0, stream>>>(
      ws + AC1, 32, ws + WB2P, 160, nullptr, ws + CU2, 160, 32, 160);
  ecc_scatter<<<ceil_div(N_EDGES * 32, TB), TB, 0, stream>>>(
      ei, e_feat, ws + CU2, 160, 0, 32, ws + AC2, 32);
  add_bias_relu<<<ceil_div(N_NODES * 32, TB), TB, 0, stream>>>(
      ws + AC2, 32, ws + CU2, 160, 128, bias2, N_NODES, 32);

  // ---- Per-graph sum pool into concatenated head input
  pool_concat<<<ceil_div(N_NODES * 32, TB), TB, 0, stream>>>(
      ws + AG2, ws + AC2, seg, ws + HC);

  // ---- MLP head (WMMA): padded B cols + relu make output pad cols zero,
  //      which is exactly the A-padding the next GEMM needs.
  gemm_wmma<1, 2><<<dim3(G_B / 32, 2), 32, 0, stream>>>(
      ws + HC, 64, ws + WD1P, 32, ws + BD1P, ws + H1, 32, 64, 32);
  gemm_wmma<1, 2><<<dim3(G_B / 32, 2), 32, 0, stream>>>(
      ws + H1, 32, ws + WD2P, 32, ws + BD2P, ws + H2, 32, 32, 32);
  gemm_wmma<2, 2><<<dim3(G_B / 32, 1), 32, 0, stream>>>(
      ws + H2, 32, ws + WOP, 16, ws + BOP, out, 1, 32, 1);
}
